// ConvTransEDecoder_429496729701
// MI455X (gfx1250) — matrix operand order repaired and checked by reading, not compiled, not run on hardware
//
#include <hip/hip_runtime.h>

// ============================================================================
// ConvTransE decoder, fused for MI455X (gfx1250, wave32, WMMA).
//
// scores[b] = sum_n ( relu(conv1d(concat(subj_e,rel_e)))_flat @ fc + fc_bias )[b,n] * obj_e[b,n]
//
// GEMM: M=8192 (batch), N=512, K=32704 (=32 filters * 1022 positions), 274 GFLOP.
// A (relu-conv) is generated on the fly per-wave in registers (never hits HBM).
// B (fc) is pre-packed to f16 K-tile-major in d_ws so staging is contiguous.
// f16 inputs / f32 accumulation via v_wmma_f32_16x16x32_f16.
// Round 4: sched_group_barrier-pinned depth-2 DS/WMMA pipeline; conv window
// read as aligned b32 (p provably even); last K-iteration peeled.
// ============================================================================

typedef _Float16 half_t;
typedef __attribute__((ext_vector_type(8)))  _Float16 v8h;
typedef __attribute__((ext_vector_type(16))) _Float16 v16h;
typedef __attribute__((ext_vector_type(8)))  float    v8f;

#define EMBED       512
#define NFILT       32
#define CONVLEN     1022          // 2*512 - 3 + 1
#define CONVK       32704         // 32 * 1022
#define NKT         1022          // K-tiles of 32
#define BK          32
#define BM          64
#define NTHREADS    512
#define COMB_STRIDE 1026          // halves per combined row (pad: 1-bank shift/row)
#define B_STRIDE    40            // halves per B row (16B aligned, conflict-free)
#define B_ROWS      512

#if __has_builtin(__builtin_amdgcn_sched_group_barrier)
#define SGB(mask, size, id) __builtin_amdgcn_sched_group_barrier(mask, size, id)
#else
#define SGB(mask, size, id)
#endif

static constexpr size_t COMB_BYTES = (size_t)BM * COMB_STRIDE * 2;      // 131328
static constexpr size_t B_BYTES    = (size_t)2 * B_ROWS * B_STRIDE * 2; // 81920
static constexpr size_t W_BYTES    = (size_t)NFILT * 4 * 4;             // 512
static constexpr size_t BIAS_BYTES = (size_t)EMBED * 4;                 // 2048
static constexpr size_t IDX_BYTES  = (size_t)BM * 4;
static constexpr size_t SMEM_TOTAL =
    COMB_BYTES + B_BYTES + W_BYTES + BIAS_BYTES + 4 * IDX_BYTES;        // ~212 KB

// ---------------------------------------------------------------------------
// Pack fc[32704][512] f32  ->  fcp[kt=1022][n=512][q=32] f16
// ---------------------------------------------------------------------------
__global__ __launch_bounds__(256) void convtranse_pack_fc(
    const float* __restrict__ fc, half_t* __restrict__ fcp)
{
  __shared__ half_t tile[128][32];   // [n][q], rows 64B -> v8h aligned
  const int kt = blockIdx.x;
  const int nb = blockIdx.y * 128;
  const int t  = threadIdx.x;

#pragma unroll
  for (int i = 0; i < 16; ++i) {
    int e = t + i * 256;             // 0..4095
    int q = e >> 7;                  // 0..31
    int n = e & 127;
    tile[n][q] = (half_t)fc[((size_t)(kt * 32 + q)) * EMBED + nb + n];
  }
  __syncthreads();

  const int n  = t >> 1;
  const int q0 = (t & 1) * 16;
  v8h* dst = (v8h*)(fcp + (((size_t)kt * B_ROWS) + nb + n) * BK + q0);
  dst[0] = *(const v8h*)&tile[n][q0];
  dst[1] = *(const v8h*)&tile[n][q0 + 8];
}

__device__ __forceinline__ v16h load_bfrag(const half_t* p) {
  const v8h* bp = (const v8h*)p;
  v8h lo = bp[0], hi = bp[2];
  return __builtin_shufflevector(lo, hi,
      0, 1, 2, 3, 4, 5, 6, 7, 8, 9, 10, 11, 12, 13, 14, 15);
}

// ---------------------------------------------------------------------------
// One K-tile: build relu-conv A fragment, 8 WMMAs with depth-2 pinned pipeline.
// ---------------------------------------------------------------------------
__device__ __forceinline__ void compute_tile(
    int kt, const half_t* __restrict__ crow, const float* __restrict__ wlds,
    const half_t* __restrict__ bwave, int halfSel, v8f (&acc)[8])
{
  // --- build A fragment (16x32 relu-conv tile, ISA lane layout) ---
  // halves 0..7  <-> K = k0 + 8*halfSel + 0..7
  // halves 8..15 <-> K = k0 + 8*halfSel + 16..23
  v16h afrag;
  const int k0 = kt * BK;
#pragma unroll
  for (int s = 0; s < 2; ++s) {
    const int kks = k0 + halfSel * 8 + s * 16;   // always even
    const unsigned f = (unsigned)kks / 1022u;
    const int p = kks - (int)f * 1022;           // always even -> 4B-aligned window
    const float w0 = wlds[f * 4 + 0], w1 = wlds[f * 4 + 1];
    const float w2 = wlds[f * 4 + 2], bb = wlds[f * 4 + 3];
    if (p <= 1014) {                   // whole 8-run inside filter f (common)
      unsigned cw[5];
#pragma unroll
      for (int j = 0; j < 5; ++j)
        cw[j] = *(const unsigned*)(crow + p + 2 * j);   // 5x ds_load_b32
      float cc[10];
#pragma unroll
      for (int j = 0; j < 5; ++j) {
        union { unsigned u; half_t h[2]; } uu; uu.u = cw[j];
        cc[2 * j]     = (float)uu.h[0];
        cc[2 * j + 1] = (float)uu.h[1];
      }
#pragma unroll
      for (int h = 0; h < 8; ++h) {
        float a = bb + w0 * cc[h] + w1 * cc[h + 1] + w2 * cc[h + 2];
        afrag[s * 8 + h] = (half_t)fmaxf(a, 0.0f);
      }
    } else {                           // run crosses a filter boundary (rare)
#pragma unroll
      for (int h = 0; h < 8; ++h) {
        const int kk = kks + h;
        const unsigned f2 = (unsigned)kk / 1022u;
        const int p2 = kk - (int)f2 * 1022;
        float a = wlds[f2 * 4 + 3]
                + wlds[f2 * 4 + 0] * (float)crow[p2]
                + wlds[f2 * 4 + 1] * (float)crow[p2 + 1]
                + wlds[f2 * 4 + 2] * (float)crow[p2 + 2];
        afrag[s * 8 + h] = (half_t)fmaxf(a, 0.0f);
      }
    }
  }

  // --- 8 B fragments + 8 WMMAs ---
  const half_t* bbase = bwave + (kt & 1) * (B_ROWS * B_STRIDE);
  v16h bf[8];
#pragma unroll
  for (int jt = 0; jt < 8; ++jt)
    bf[jt] = load_bfrag(bbase + (jt * 4 * 16) * B_STRIDE);
#pragma unroll
  for (int jt = 0; jt < 8; ++jt)
    acc[jt] = __builtin_amdgcn_wmma_f32_16x16x32_f16(
        false, afrag, false, bf[jt], (short)0, acc[jt], false, false);

  // --- pin a depth-2 DS-read / WMMA pipeline:
  //     DSx4, (WMMAx1, DSx2)x6, WMMAx2  (0x100 = DS read, 0x8 = MFMA/WMMA)
  SGB(0x100, 4, 0);
#pragma unroll
  for (int x = 0; x < 6; ++x) {
    SGB(0x008, 1, 0);
    SGB(0x100, 2, 0);
  }
  SGB(0x008, 2, 0);
}

// ---------------------------------------------------------------------------
// Fused gather + conv + relu + GEMM(WMMA) + bias + dot(obj) kernel.
// One block = 64 batch rows x all 512 output cols. 16 waves = 4(M) x 4(N).
// ---------------------------------------------------------------------------
__global__ __launch_bounds__(NTHREADS) void convtranse_main(
    const float* __restrict__ entity, const float* __restrict__ rel,
    const float* __restrict__ convw,  const float* __restrict__ convb,
    const float* __restrict__ fcbias, const long long* __restrict__ triples,
    const half_t* __restrict__ fcp,   float* __restrict__ out)
{
  extern __shared__ char smem[];
  half_t* comb    = (half_t*)smem;                                    // [64][1026] f16
  half_t* Blds    = (half_t*)(smem + COMB_BYTES);                     // [2][512][40] f16
  float*  wlds    = (float*)(smem + COMB_BYTES + B_BYTES);            // [32][4] (w0,w1,w2,bias)
  float*  biaslds = wlds + NFILT * 4;                                 // [512]
  int*    objIdx  = (int*)(biaslds + EMBED);                          // [64]
  int*    subjIdx = objIdx + BM;                                      // [64]
  int*    relIdx  = subjIdx + BM;                                     // [64]
  float*  scores  = (float*)(relIdx + BM);                            // [64]

  const int t     = threadIdx.x;
  const int mBase = blockIdx.x * BM;

  // --- stage indices, conv weights, fc_bias, zero scores ---
  if (t < BM) {
    const long long* tr = triples + (size_t)(mBase + t) * 3;
    subjIdx[t] = (int)tr[0];
    relIdx[t]  = (int)tr[1];
    objIdx[t]  = (int)tr[2];
    scores[t]  = 0.0f;
  }
  if (t < NFILT) {
    wlds[t * 4 + 0] = convw[t * 3 + 0];
    wlds[t * 4 + 1] = convw[t * 3 + 1];
    wlds[t * 4 + 2] = convw[t * 3 + 2];
    wlds[t * 4 + 3] = convb[t];
  }
  biaslds[t] = fcbias[t];            // NTHREADS == EMBED == 512
  __syncthreads();

  // --- stage combined[64][1024] as f16 into LDS (gather subj_e | rel_e) ---
#pragma unroll
  for (int i = 0; i < 32; ++i) {
    int e   = t + i * NTHREADS;      // 0..16383 float4 units
    int row = e >> 8;                // 256 float4 per row
    int col = (e & 255) * 4;
    const float* src = (col < EMBED)
        ? entity + (size_t)subjIdx[row] * EMBED + col
        : rel    + (size_t)relIdx[row]  * EMBED + (col - EMBED);
    float4 v = *(const float4*)src;
    half_t* d = comb + row * COMB_STRIDE + col;
    union { half_t h[2]; unsigned u; } p0, p1;
    p0.h[0] = (half_t)v.x; p0.h[1] = (half_t)v.y;
    p1.h[0] = (half_t)v.z; p1.h[1] = (half_t)v.w;
    *(unsigned*)(d)     = p0.u;
    *(unsigned*)(d + 2) = p1.u;
  }

  // --- stage B tile 0 (thread t copies fcp row t: 32 halves = 64B) ---
  {
    const v8h* src = (const v8h*)(fcp + (size_t)t * BK);
    v8h* dst = (v8h*)(Blds + t * B_STRIDE);
    dst[0] = src[0]; dst[1] = src[1]; dst[2] = src[2]; dst[3] = src[3];
  }
  __syncthreads();

  // --- wave tiling ---
  const int wave    = t >> 5;
  const int lane    = t & 31;
  const int mi      = wave >> 2;       // 0..3 : M-tile
  const int ni      = wave & 3;        // 0..3 : N phase (tiles ni, ni+4, ..., ni+28)
  const int mb      = mi * 16;
  const int halfSel = lane >> 4;       // 0: lanes 0-15, 1: lanes 16-31
  const int l16     = lane & 15;
  const half_t* crow  = comb + (mb + l16) * COMB_STRIDE;
  const half_t* bwave = Blds + halfSel * 8 + (ni * 16 + l16) * B_STRIDE;

  v8f acc[8] = {};

  // ============== K loop (last iteration peeled: no branches inside) ==============
  for (int kt = 0; kt < NKT - 1; ++kt) {
    // prefetch next B tile into registers (overlaps with compute)
    const v8h* src = (const v8h*)(fcp + ((size_t)(kt + 1) * B_ROWS + t) * BK);
    v8h pf0 = src[0], pf1 = src[1], pf2 = src[2], pf3 = src[3];

    compute_tile(kt, crow, wlds, bwave, halfSel, acc);

    v8h* dst = (v8h*)(Blds + ((kt + 1) & 1) * (B_ROWS * B_STRIDE) + t * B_STRIDE);
    dst[0] = pf0; dst[1] = pf1; dst[2] = pf2; dst[3] = pf3;
    __syncthreads();
  }
  compute_tile(NKT - 1, crow, wlds, bwave, halfSel, acc);

  // ====================== endgame: (+bias) dot obj_e ======================
  // C layout: VGPR r, lanes 0-15 -> M=r, lanes 16-31 -> M=r+8, N = lane%16.
#pragma unroll
  for (int jt = 0; jt < 8; ++jt) {
    const int n = (ni + jt * 4) * 16 + l16;
    const float fb = biaslds[n];
#pragma unroll
    for (int r = 0; r < 8; ++r) {
      const int mLoc = mb + r + halfSel * 8;
      const float* orow = entity + (size_t)objIdx[mLoc] * EMBED;
      float val = (acc[jt][r] + fb) * orow[n];
      val += __shfl_xor(val, 8, 16);
      val += __shfl_xor(val, 4, 16);
      val += __shfl_xor(val, 2, 16);
      val += __shfl_xor(val, 1, 16);
      if (l16 == 0) atomicAdd(&scores[mLoc], val);
    }
  }
  __syncthreads();
  if (t < BM) out[mBase + t] = scores[t];
}

// ---------------------------------------------------------------------------
extern "C" void kernel_launch(void* const* d_in, const int* in_sizes, int n_in,
                              void* d_out, int out_size, void* d_ws, size_t ws_size,
                              hipStream_t stream) {
  const float*     entity  = (const float*)d_in[0];
  const float*     rel     = (const float*)d_in[1];
  const float*     convw   = (const float*)d_in[2];
  const float*     convb   = (const float*)d_in[3];
  const float*     fc      = (const float*)d_in[4];
  const float*     fcbias  = (const float*)d_in[5];
  const long long* triples = (const long long*)d_in[6];
  float*           outp    = (float*)d_out;

  half_t* fcp = (half_t*)d_ws;     // 1022*512*32 f16 = ~32 MB

  dim3 gridP(NKT, 4);
  convtranse_pack_fc<<<gridP, 256, 0, stream>>>(fc, fcp);

  convtranse_main<<<8192 / BM, NTHREADS, SMEM_TOTAL, stream>>>(
      entity, rel, convw, convb, fcbias, triples, fcp, outp);
}
